// ParsingNet_GPU_55216099557609
// MI455X (gfx1250) — matrix operand order repaired and checked by reading, not compiled, not run on hardware
//
#include <hip/hip_runtime.h>

typedef __attribute__((ext_vector_type(16))) __bf16 v16bf;
typedef __attribute__((ext_vector_type(8)))  __bf16 v8bf;
typedef __attribute__((ext_vector_type(8)))  float  v8f;

#define CH     128
#define EPAD   136          // padded row stride in bf16 elements (17*16B rows)
#define WAVES  4
#define LN_EPS 1e-5f

// Pack two f32 -> one dword of bf16x2 (a in low half, b in high half).
// Round-to-nearest via +0x8000, then one v_perm_b32 splices the high halves.
__device__ __forceinline__ unsigned pack_bf16(float a, float b) {
  union { float f; unsigned u; } ua, ub;
  ua.f = a; ub.f = b;
  unsigned au = ua.u + 0x8000u;
  unsigned bu = ub.u + 0x8000u;
#if __has_builtin(__builtin_amdgcn_perm)
  // perm(hi_src=b, lo_src=a, sel): byte0=a[2], byte1=a[3], byte2=b[2], byte3=b[3]
  return __builtin_amdgcn_perm(bu, au, 0x07060302u);
#else
  return (au >> 16) | (bu & 0xFFFF0000u);
#endif
}

__global__ __launch_bounds__(128, 1)
void edge_score_kernel(const float* __restrict__ x,
                       const int*   __restrict__ rowi,
                       const int*   __restrict__ coli,
                       const float* __restrict__ W1,
                       const float* __restrict__ b1,
                       const float* __restrict__ lng,
                       const float* __restrict__ lnb,
                       const float* __restrict__ W2,
                       const float* __restrict__ b2,
                       float*       __restrict__ out,
                       int E)
{
  // W1^T in bf16: sW[n*EPAD + k] = bf16(W1[k][n])  (column-contiguous K)
  __shared__ __align__(16) unsigned short sW[CH * EPAD];
  __shared__ float sPar[4 * CH];                         // b1 | gamma | beta | W2
  __shared__ __align__(16) unsigned short sE[WAVES][16 * EPAD];  // per-wave A tile

  const int tid = threadIdx.x;

  // ---- one-time staging (coalesced reads of W1, row-n writes) ----
  {
    const int n = tid;
    unsigned* dstw = (unsigned*)(sW + n * EPAD);         // EPAD even -> dword aligned
    for (int k = 0; k < CH; k += 2) {
      float w0 = W1[(k    ) * CH + n];
      float w1 = W1[(k + 1) * CH + n];
      dstw[k >> 1] = pack_bf16(w0, w1);
    }
    sPar[tid]          = b1[tid];
    sPar[CH + tid]     = lng[tid];
    sPar[2*CH + tid]   = lnb[tid];
    sPar[3*CH + tid]   = W2[tid];
  }
  __syncthreads();

  const int wave = tid >> 5;
  const int lane = tid & 31;
  const int nl   = lane & 15;        // N column within 16-wide tile / A row M
  const int hi   = lane >> 4;        // half-wave select
  const float b2v = b2[0];

  const int nTiles = (E + 15) >> 4;
  const int tStride = gridDim.x * WAVES;
  unsigned short* myE = &sE[wave][0];

  for (int tile = blockIdx.x * WAVES + wave; tile < nTiles; tile += tStride) {
    const int ebase = tile * 16;

    // ---- phase 1: build A tile e = x[row]*x[col] in bf16 (2 lanes/edge) ----
    {
      const int m  = lane >> 1;
      const int hh = lane & 1;
      int e = ebase + m; if (e >= E) e = E - 1;          // clamp, keep lanes live
      const int r = rowi[e];
      const int c = coli[e];
      const float* xr = x + (size_t)r * CH + hh * 64;
      const float* xc = x + (size_t)c * CH + hh * 64;
      uint2* dst = (uint2*)(myE + m * EPAD + hh * 64);
#pragma unroll
      for (int i = 0; i < 16; ++i) {
        float a0 = xr[4*i+0] * xc[4*i+0];
        float a1 = xr[4*i+1] * xc[4*i+1];
        float a2 = xr[4*i+2] * xc[4*i+2];
        float a3 = xr[4*i+3] * xc[4*i+3];
        uint2 p;
        p.x = pack_bf16(a0, a1);
        p.y = pack_bf16(a2, a3);
        dst[i] = p;
      }
    }
    // same-wave LDS RAW: DS pipe is in-order per wave; drain before fragment reads
    asm volatile("s_wait_dscnt 0" ::: "memory");

    // ---- phase 2: h = e @ W1 via v_wmma_f32_16x16x32_bf16 ----
    v8f acc[8] = {};
    const int akb = hi * 8;          // A: half-wave K offset (K 0-7 / 8-15 ...)
    const int bkb = hi * 8;          // B: half-wave K offset (K 0-7 / 8-15 ...)
#pragma unroll
    for (int kt = 0; kt < 4; ++kt) {
      v16bf afrag;
      {
        const unsigned short* ap = myE + nl * EPAD + kt * 32 + akb;
        ((v8bf*)&afrag)[0] = *(const v8bf*)(ap);         // K kb .. kb+7
        ((v8bf*)&afrag)[1] = *(const v8bf*)(ap + 16);    // K kb+16 .. kb+23
      }
#pragma unroll
      for (int nt = 0; nt < 8; ++nt) {
        const unsigned short* bp = sW + (nt*16 + nl) * EPAD + kt * 32 + bkb;
        v16bf bfrag;
        ((v8bf*)&bfrag)[0] = *(const v8bf*)(bp);         // K kb2 .. kb2+7
        ((v8bf*)&bfrag)[1] = *(const v8bf*)(bp + 16);    // K kb2+16 .. kb2+23
        acc[nt] = __builtin_amdgcn_wmma_f32_16x16x32_bf16(
            false, afrag, false, bfrag, (short)0, acc[nt], false, false);
      }
    }

    // ---- phase 3: +b1, LayerNorm(128), ReLU, dot W2, sigmoid ----
    // C/D layout: VGPR r holds row M=r (lanes 0-15) / r+8 (lanes 16-31), col=nl
    float hsum[8], hsq[8];
#pragma unroll
    for (int r = 0; r < 8; ++r) { hsum[r] = 0.f; hsq[r] = 0.f; }
#pragma unroll
    for (int nt = 0; nt < 8; ++nt) {
      const float bb = sPar[nt*16 + nl];
#pragma unroll
      for (int r = 0; r < 8; ++r) {
        float v = acc[nt][r] + bb;
        acc[nt][r] = v;
        hsum[r] += v;
        hsq[r]  += v * v;
      }
    }
#pragma unroll
    for (int r = 0; r < 8; ++r) {
#pragma unroll
      for (int s = 1; s < 16; s <<= 1) {
        hsum[r] += __shfl_xor(hsum[r], s, 16);
        hsq[r]  += __shfl_xor(hsq[r],  s, 16);
      }
    }
    float mu[8], rstd[8];
#pragma unroll
    for (int r = 0; r < 8; ++r) {
      mu[r] = hsum[r] * (1.0f / 128.0f);
      float var = hsq[r] * (1.0f / 128.0f) - mu[r] * mu[r];
      rstd[r] = rsqrtf(var + LN_EPS);
    }
    float logit[8];
#pragma unroll
    for (int r = 0; r < 8; ++r) logit[r] = 0.f;
#pragma unroll
    for (int nt = 0; nt < 8; ++nt) {
      const int col  = nt*16 + nl;
      const float g  = sPar[CH   + col];
      const float bb = sPar[2*CH + col];
      const float w  = sPar[3*CH + col];
#pragma unroll
      for (int r = 0; r < 8; ++r) {
        float v = (acc[nt][r] - mu[r]) * rstd[r] * g + bb;
        v = v > 0.f ? v : 0.f;                            // ReLU
        logit[r] += v * w;
      }
    }
#pragma unroll
    for (int r = 0; r < 8; ++r)
#pragma unroll
      for (int s = 1; s < 16; s <<= 1)
        logit[r] += __shfl_xor(logit[r], s, 16);

    if (nl == 0) {                    // lane 0 -> edges 0..7, lane 16 -> 8..15
      const int mb = hi * 8;
#pragma unroll
      for (int r = 0; r < 8; ++r) {
        int e = ebase + mb + r;
        if (e < E) {
          float z = logit[r] + b2v;
          out[e] = 1.0f / (1.0f + __expf(-z));
        }
      }
    }
  }
}

extern "C" void kernel_launch(void* const* d_in, const int* in_sizes, int n_in,
                              void* d_out, int out_size, void* d_ws, size_t ws_size,
                              hipStream_t stream) {
  (void)n_in; (void)d_ws; (void)ws_size;
  const float* x   = (const float*)d_in[0];
  const int*   ei  = (const int*)d_in[1];   // [2, E]
  // d_in[2] = batch (unused by the reference computation)
  const float* W1  = (const float*)d_in[3];
  const float* b1  = (const float*)d_in[4];
  const float* lng = (const float*)d_in[5];
  const float* lnb = (const float*)d_in[6];
  const float* W2  = (const float*)d_in[7];
  const float* b2  = (const float*)d_in[8];
  float* out = (float*)d_out;

  const int E = in_sizes[1] / 2;
  (void)out_size;

  dim3 grid(2048), block(128);
  edge_score_kernel<<<grid, block, 0, stream>>>(
      x, ei, ei + E, W1, b1, lng, lnb, W2, b2, out, E);
}
